// MultiheadAttention_69423851372602
// MI455X (gfx1250) — matrix operand order repaired and checked by reading, not compiled, not run on hardware
//
#include <hip/hip_runtime.h>
#include <hip/hip_bf16.h>

// ---------------------------------------------------------------------------
// MHA forward for MI455X (gfx1250, wave32, WMMA 16x16x32 f16).
// B=4, S=1024, D=1024, H=16, HD=64.  All GEMMs run on v_wmma_f32_16x16x32_f16.
// Double-buffered LDS tiles (1 barrier / step), global_prefetch for the
// tile after next, templated epilogue (no runtime branches).
// Flash attention processes 64 keys per iteration (softmax overhead halved
// per key vs 32-key tiles).
// ---------------------------------------------------------------------------

typedef __attribute__((ext_vector_type(16))) _Float16 v16h;
typedef __attribute__((ext_vector_type(8)))  float    v8f;

union HPair { unsigned u; _Float16 h[2]; };

__device__ inline v8f v8f_zero() {
  v8f z;
#pragma unroll
  for (int i = 0; i < 8; ++i) z[i] = 0.0f;
  return z;
}

static constexpr int Bsz = 4;
static constexpr int S   = 1024;
static constexpr int D   = 1024;
static constexpr int H   = 16;
static constexpr int HD  = 64;
static constexpr int Mrows = Bsz * S;        // 4096

// ---------------------------------------------------------------------------
// fp32 -> f16 cast, vectorized (float4 in, b64 out)
// ---------------------------------------------------------------------------
__global__ void cast_f32_to_f16(const float* __restrict__ src,
                                _Float16* __restrict__ dst, int n) {
  int i = blockIdx.x * blockDim.x + threadIdx.x;
  int stride = gridDim.x * blockDim.x;
  int n4 = n >> 2;  // n is always a multiple of 4 here
  for (int j = i; j < n4; j += stride) {
    const float4 f = ((const float4*)src)[j];
    union { _Float16 h[4]; unsigned long long u; } p;
    p.h[0] = (_Float16)f.x;
    p.h[1] = (_Float16)f.y;
    p.h[2] = (_Float16)f.z;
    p.h[3] = (_Float16)f.w;
    *(unsigned long long*)(dst + 4 * (size_t)j) = p.u;
  }
}

// ---------------------------------------------------------------------------
// Tiled WMMA GEMM:  Y = act(A[4096,1024] @ W[1024,1024] + bias)
//   RELU_MODE==1 : Y = relu(...), stored f16 into outH
//   RELU_MODE==0 : Y = ... + bias, stored f32 into outF
// Workgroup: 256 threads (8 waves), tile 128x128.
// Wave grid 4(row) x 2(col); each wave -> 32x64 = 2x4 fragments of 16x16.
// W tile (32 x 128) staged transposed in LDS (double-buffered) so
// B-fragments are contiguous dword ds reads; A-fragments come straight from
// global as b128 loads.
// ---------------------------------------------------------------------------
template <int RELU_MODE>
__global__ __launch_bounds__(256) void wmma_gemm_kernel(
    const _Float16* __restrict__ A, const _Float16* __restrict__ W,
    const float* __restrict__ bias, _Float16* __restrict__ outH,
    float* __restrict__ outF) {
  constexpr int Kdim = D;
  constexpr int N = D;

  const int tid  = threadIdx.x;
  const int lane = tid & 31;
  const int wave = tid >> 5;
  const int n16  = lane & 15;
  const int hl   = lane >> 4;

  const int waveRow = wave & 3;   // 0..3  -> 32 rows each
  const int waveCol = wave >> 2;  // 0..1  -> 64 cols each
  const int tileM = blockIdx.y * 128;
  const int tileN = blockIdx.x * 128;
  const int mBase = tileM + waveRow * 32;
  const int nBase = tileN + waveCol * 64;

  __shared__ _Float16 BT[2][128][34];  // W tile transposed: [n][k], +2 pad

  // Cooperative W tile staging: 32x128 f16 = 2048 dwords, 8 dwords/thread.
  unsigned wreg[8];
  auto fetch_w = [&](int k0) {
#pragma unroll
    for (int it = 0; it < 8; ++it) {
      int j  = tid + it * 256;   // 0..2047
      int kk = j >> 6;           // 0..31
      int nn = (j & 63) * 2;     // 0..126
      wreg[it] = *(const unsigned*)(W + (size_t)(k0 + kk) * N + tileN + nn);
    }
  };
  auto store_w = [&](int buf) {
#pragma unroll
    for (int it = 0; it < 8; ++it) {
      int j  = tid + it * 256;
      int kk = j >> 6;
      int nn = (j & 63) * 2;
      HPair u; u.u = wreg[it];
      BT[buf][nn][kk]     = u.h[0];
      BT[buf][nn + 1][kk] = u.h[1];
    }
  };

  v8f acc[2][4];
#pragma unroll
  for (int mb = 0; mb < 2; ++mb)
#pragma unroll
    for (int nb = 0; nb < 4; ++nb) acc[mb][nb] = v8f_zero();

  fetch_w(0);
  store_w(0);
  __syncthreads();

  for (int k0 = 0; k0 < Kdim; k0 += 32) {
    const int cur = (k0 >> 5) & 1;
    const bool hasNext = (k0 + 32) < Kdim;

    if (hasNext) fetch_w(k0 + 32);          // overlap with WMMAs below
    if (k0 + 64 < Kdim)                     // prefetch tile after next
      __builtin_prefetch(
          W + (size_t)(k0 + 64 + (tid >> 3)) * N + tileN + (tid & 7) * 16, 0, 0);

    // A fragments (16x32 f16) straight from global: contiguous dword pairs.
    v16h af[2];
#pragma unroll
    for (int mb = 0; mb < 2; ++mb) {
      const _Float16* ap = A + (size_t)(mBase + mb * 16 + n16) * Kdim + k0;
#pragma unroll
      for (int v = 0; v < 8; ++v) {
        int k = ((v & 4) ? 16 : 0) + hl * 8 + (v & 3) * 2;
        HPair u; u.u = *(const unsigned*)(ap + k);
        af[mb][2 * v]     = u.h[0];
        af[mb][2 * v + 1] = u.h[1];
      }
    }

#pragma unroll
    for (int nb = 0; nb < 4; ++nb) {
      v16h bf;
      int nl = waveCol * 64 + nb * 16 + n16;  // column within 128-tile
#pragma unroll
      for (int v = 0; v < 8; ++v) {
        int k = ((v & 4) ? 16 : 0) + hl * 8 + (v & 3) * 2;
        HPair u; u.u = *(const unsigned*)&BT[cur][nl][k];
        bf[2 * v]     = u.h[0];
        bf[2 * v + 1] = u.h[1];
      }
#pragma unroll
      for (int mb = 0; mb < 2; ++mb)
        acc[mb][nb] = __builtin_amdgcn_wmma_f32_16x16x32_f16(
            false, af[mb], false, bf, (short)0, acc[mb][nb], false, false);
    }

    if (hasNext) store_w(cur ^ 1);          // fill the other buffer
    __syncthreads();                        // one barrier per k-step
  }

  // Straight-line epilogue (template-resolved mode).
#pragma unroll
  for (int nb = 0; nb < 4; ++nb) {
    int col = nBase + nb * 16 + n16;
    float bcol = bias[col];
#pragma unroll
    for (int mb = 0; mb < 2; ++mb) {
#pragma unroll
      for (int v = 0; v < 8; ++v) {
        int row = mBase + mb * 16 + v + 8 * hl;
        float val = acc[mb][nb][v] + bcol;
        if (RELU_MODE) {
          val = fmaxf(val, 0.0f);
          outH[(size_t)row * N + col] = (_Float16)val;
        } else {
          outF[(size_t)row * N + col] = val;
        }
      }
    }
  }
}

// ---------------------------------------------------------------------------
// Flash-attention over f16 Q/K/V (post-projection), f16 output.
// Grid: (S/128, H, B). Block: 256 threads = 8 waves; wave w owns q-rows
// [blk*128 + w*16, +16) of head (b,h).  64-key tiles per iteration:
//   scores: 4 16x16 fragments (2 WMMAs each, K=32 over d), scale
//   1/sqrt(D)=1/32 folded into Q fragments; one online-softmax row-max
//   butterfly + one accumulator rescale per 64 keys; P (16x64) bounced
//   through per-wave LDS (C-layout -> A-layout, 2 fragments); PV: 2x4
//   accumulate WMMAs with V tile staged transposed in LDS (double-buffered,
//   1 barrier / tile).
// ---------------------------------------------------------------------------
__global__ __launch_bounds__(256) void flash_attn_kernel(
    const _Float16* __restrict__ Q, const _Float16* __restrict__ K,
    const _Float16* __restrict__ V, _Float16* __restrict__ Oattn) {
  const int tid  = threadIdx.x;
  const int lane = tid & 31;
  const int wave = tid >> 5;
  const int n16  = lane & 15;
  const int hl   = lane >> 4;

  const int qBlk = blockIdx.x;            // 0..7
  const int h    = blockIdx.y;            // 0..15
  const int b    = blockIdx.z;            // 0..3

  const int rowBase = b * S;              // row offset into [4096, D]
  const int qRow0   = qBlk * 128 + wave * 16;
  const int colBase = h * HD;

  __shared__ _Float16 VT[2][64][66];      // V tile transposed [d][t], +2 pad
  __shared__ _Float16 Pbuf[8][16][64];    // per-wave P tile, row-major [m][t]

  // Cooperative V tile staging: 64 t x 64 d, 8 dwords (16 f16) per thread.
  const int tl = tid >> 2;                // 0..63 : t within tile
  const int dg = (tid & 3) * 16;          // 0/16/32/48 : d group
  unsigned vreg[8];
  auto fetch_v = [&](int t0) {
    const _Float16* vp = V + (size_t)(rowBase + t0 + tl) * D + colBase + dg;
#pragma unroll
    for (int j = 0; j < 8; ++j) vreg[j] = *(const unsigned*)(vp + 2 * j);
  };
  auto store_v = [&](int buf) {
#pragma unroll
    for (int j = 0; j < 8; ++j) {
      HPair u; u.u = vreg[j];
      VT[buf][dg + 2 * j][tl]     = u.h[0];
      VT[buf][dg + 2 * j + 1][tl] = u.h[1];
    }
  };

  // Q fragments, resident all loop long; scale = 1/sqrt(1024) = 1/32 folded in.
  v16h qf[2];
#pragma unroll
  for (int dc = 0; dc < 2; ++dc) {
    const _Float16* qp =
        Q + (size_t)(rowBase + qRow0 + n16) * D + colBase + dc * 32;
#pragma unroll
    for (int v = 0; v < 8; ++v) {
      int k = ((v & 4) ? 16 : 0) + hl * 8 + (v & 3) * 2;
      HPair u; u.u = *(const unsigned*)(qp + k);
      qf[dc][2 * v]     = u.h[0] * (_Float16)0.03125f;
      qf[dc][2 * v + 1] = u.h[1] * (_Float16)0.03125f;
    }
  }

  v8f acc[4];
#pragma unroll
  for (int f = 0; f < 4; ++f) acc[f] = v8f_zero();
  float rowmax[8], rowsum[8];
#pragma unroll
  for (int v = 0; v < 8; ++v) { rowmax[v] = -3.0e38f; rowsum[v] = 0.0f; }

  fetch_v(0);
  store_v(0);
  __syncthreads();

  for (int t0 = 0; t0 < S; t0 += 64) {
    const int cur = (t0 >> 6) & 1;
    const bool hasNext = (t0 + 64) < S;

    if (hasNext) {
      fetch_v(t0 + 64);  // overlap with score WMMAs + softmax below
      __builtin_prefetch(
          K + (size_t)(rowBase + t0 + 64 + (tid >> 2)) * D + colBase, 0, 0);
    }

    // Scores: four 16x16 fragments over this 64-key tile.
    v8f sc[4];
#pragma unroll
    for (int tb = 0; tb < 4; ++tb) {
      v8f s = v8f_zero();
#pragma unroll
      for (int dc = 0; dc < 2; ++dc) {
        v16h bk;
        const _Float16* kp = K +
            (size_t)(rowBase + t0 + tb * 16 + n16) * D + colBase + dc * 32;
#pragma unroll
        for (int v = 0; v < 8; ++v) {
          int k = ((v & 4) ? 16 : 0) + hl * 8 + (v & 3) * 2;
          HPair u; u.u = *(const unsigned*)(kp + k);
          bk[2 * v]     = u.h[0];
          bk[2 * v + 1] = u.h[1];
        }
        s = __builtin_amdgcn_wmma_f32_16x16x32_f16(
            false, qf[dc], false, bk, (short)0, s, false, false);
      }
      sc[tb] = s;
    }

    // Online softmax: one butterfly + one rescale per 64 keys.
    // Row m = v + 8*hl lives in the 16-lane half; reduce across n with
    // shfl_xor masks 1,2,4,8.
#pragma unroll
    for (int v = 0; v < 8; ++v) {
      float t = fmaxf(fmaxf(sc[0][v], sc[1][v]), fmaxf(sc[2][v], sc[3][v]));
      t = fmaxf(t, __shfl_xor(t, 1, 32));
      t = fmaxf(t, __shfl_xor(t, 2, 32));
      t = fmaxf(t, __shfl_xor(t, 4, 32));
      t = fmaxf(t, __shfl_xor(t, 8, 32));
      float nm = fmaxf(rowmax[v], t);
      float alpha = __expf(rowmax[v] - nm);
      rowmax[v] = nm;
      rowsum[v] *= alpha;
#pragma unroll
      for (int f = 0; f < 4; ++f) acc[f][v] *= alpha;
      float psum = 0.0f;
#pragma unroll
      for (int tb = 0; tb < 4; ++tb) {
        float p = __expf(sc[tb][v] - nm);
        psum += p;
        Pbuf[wave][v + 8 * hl][tb * 16 + n16] = (_Float16)p;
      }
      rowsum[v] += psum;                   // partial over this lane's n
    }

    // PV accumulate: 2 k-chunks of 32 keys x 4 d-fragments.
#pragma unroll
    for (int kc = 0; kc < 2; ++kc) {
      // Reload P chunk as an A-fragment (16x32 f16) from per-wave LDS.
      v16h pa;
#pragma unroll
      for (int v = 0; v < 8; ++v) {
        int k = kc * 32 + ((v & 4) ? 16 : 0) + hl * 8 + (v & 3) * 2;
        HPair u; u.u = *(const unsigned*)&Pbuf[wave][n16][k];
        pa[2 * v]     = u.h[0];
        pa[2 * v + 1] = u.h[1];
      }
#pragma unroll
      for (int f = 0; f < 4; ++f) {
        v16h bv;
#pragma unroll
        for (int v = 0; v < 8; ++v) {
          int t = kc * 32 + ((v & 4) ? 16 : 0) + hl * 8 + (v & 3) * 2;
          HPair u; u.u = *(const unsigned*)&VT[cur][f * 16 + n16][t];
          bv[2 * v]     = u.h[0];
          bv[2 * v + 1] = u.h[1];
        }
        acc[f] = __builtin_amdgcn_wmma_f32_16x16x32_f16(
            false, pa, false, bv, (short)0, acc[f], false, false);
      }
    }

    if (hasNext) store_v(cur ^ 1);
    __syncthreads();                        // one barrier per 64-key tile
  }

  // Normalize rows and write f16 output [4096, D] (heads re-merged).
  float inv[8];
#pragma unroll
  for (int v = 0; v < 8; ++v) {
    float s = rowsum[v];
    s += __shfl_xor(s, 1, 32);
    s += __shfl_xor(s, 2, 32);
    s += __shfl_xor(s, 4, 32);
    s += __shfl_xor(s, 8, 32);
    inv[v] = 1.0f / s;
  }
#pragma unroll
  for (int f = 0; f < 4; ++f) {
    int col = colBase + f * 16 + n16;
#pragma unroll
    for (int v = 0; v < 8; ++v) {
      int row = rowBase + qRow0 + v + 8 * hl;
      Oattn[(size_t)row * D + col] = (_Float16)(acc[f][v] * inv[v]);
    }
  }
}

// ---------------------------------------------------------------------------
// Host-side orchestration.
// d_in order: q,k,v, Wq,bq, Wk,bk, Wv,bv, Wo,bo
// Workspace layout (f16 buffers), total 64 MiB:
//   q16 k16 v16 : 8 MiB each @ 0 / 8 / 16 MiB
//   Wq16 Wk16 Wv16 Wo16 : 2 MiB each @ 24 / 26 / 28 / 30 MiB
//   Q16 K16 V16 : 8 MiB each @ 32 / 40 / 48 MiB
//   attn16 : 8 MiB @ 56 MiB
// ---------------------------------------------------------------------------
extern "C" void kernel_launch(void* const* d_in, const int* in_sizes, int n_in,
                              void* d_out, int out_size, void* d_ws,
                              size_t ws_size, hipStream_t stream) {
  const float* q  = (const float*)d_in[0];
  const float* k  = (const float*)d_in[1];
  const float* v  = (const float*)d_in[2];
  const float* Wq = (const float*)d_in[3];
  const float* bq = (const float*)d_in[4];
  const float* Wk = (const float*)d_in[5];
  const float* bk = (const float*)d_in[6];
  const float* Wv = (const float*)d_in[7];
  const float* bv = (const float*)d_in[8];
  const float* Wo = (const float*)d_in[9];
  const float* bo = (const float*)d_in[10];
  float* out = (float*)d_out;

  char* ws = (char*)d_ws;
  const size_t MB = 1024 * 1024;
  _Float16* q16    = (_Float16*)(ws + 0 * MB);
  _Float16* k16    = (_Float16*)(ws + 8 * MB);
  _Float16* v16    = (_Float16*)(ws + 16 * MB);
  _Float16* Wq16   = (_Float16*)(ws + 24 * MB);
  _Float16* Wk16   = (_Float16*)(ws + 26 * MB);
  _Float16* Wv16   = (_Float16*)(ws + 28 * MB);
  _Float16* Wo16   = (_Float16*)(ws + 30 * MB);
  _Float16* Q16    = (_Float16*)(ws + 32 * MB);
  _Float16* K16    = (_Float16*)(ws + 40 * MB);
  _Float16* V16    = (_Float16*)(ws + 48 * MB);
  _Float16* attn16 = (_Float16*)(ws + 56 * MB);

  const int nAct = Mrows * D;   // 4 Mi elems
  const int nW   = D * D;       // 1 Mi elems

  cast_f32_to_f16<<<1024, 256, 0, stream>>>(q, q16, nAct);
  cast_f32_to_f16<<<1024, 256, 0, stream>>>(k, k16, nAct);
  cast_f32_to_f16<<<1024, 256, 0, stream>>>(v, v16, nAct);
  cast_f32_to_f16<<<512, 256, 0, stream>>>(Wq, Wq16, nW);
  cast_f32_to_f16<<<512, 256, 0, stream>>>(Wk, Wk16, nW);
  cast_f32_to_f16<<<512, 256, 0, stream>>>(Wv, Wv16, nW);
  cast_f32_to_f16<<<512, 256, 0, stream>>>(Wo, Wo16, nW);

  dim3 gblock(256);
  dim3 ggrid(D / 128, Mrows / 128);  // (8, 32)
  wmma_gemm_kernel<1><<<ggrid, gblock, 0, stream>>>(q16, Wq16, bq, Q16, nullptr);
  wmma_gemm_kernel<1><<<ggrid, gblock, 0, stream>>>(k16, Wk16, bk, K16, nullptr);
  wmma_gemm_kernel<1><<<ggrid, gblock, 0, stream>>>(v16, Wv16, bv, V16, nullptr);

  dim3 agrid(S / 128, H, Bsz);       // (8, 16, 4)
  flash_attn_kernel<<<agrid, gblock, 0, stream>>>(Q16, K16, V16, attn16);

  wmma_gemm_kernel<0><<<ggrid, gblock, 0, stream>>>(attn16, Wo16, bo, nullptr, out);
}
